// WeightOnlyInt8Linear_20916490731809
// MI455X (gfx1250) — matrix operand order repaired and checked by reading, not compiled, not run on hardware
//
#include <hip/hip_runtime.h>
#include <stdint.h>

// ---------------------------------------------------------------------------
// Weight-only int8 linear: out = (x @ W^T) * scales
//   x: [256, 4096] f32, W: [16384, 4096] int8, scales: [16384] f32
// f16 WMMA GEMM with f32 accumulation; W is exact in f16.
// ---------------------------------------------------------------------------

typedef __attribute__((ext_vector_type(16))) _Float16 v16h;
typedef __attribute__((ext_vector_type(8)))  float    v8f;
typedef int v4i_b __attribute__((__vector_size__(4 * sizeof(int))));

#define GEMM_M 256
#define GEMM_N 16384
#define GEMM_K 4096
#define BM 128
#define BN 128
#define KC 32
#define A_STRIDE 36   // dwords: 32 f32 + 4 pad  -> 4n mod 64 bank starts, conflict-free b128
#define B_STRIDE 48   // bytes : 32 i8 + 16 pad -> 12n dword starts mod 64, conflict-free b128

#if defined(__HIP_DEVICE_COMPILE__) && __has_builtin(__builtin_amdgcn_global_load_async_to_lds_b128)
#define HAS_ASYNC_LDS 1
#else
#define HAS_ASYNC_LDS 0
#endif

__device__ __forceinline__ void cp_b128(const void* g, void* l) {
#if HAS_ASYNC_LDS
  __builtin_amdgcn_global_load_async_to_lds_b128(
      (__attribute__((address_space(1))) v4i_b*)(void*)(g),
      (__attribute__((address_space(3))) v4i_b*)(l),
      /*offset=*/0, /*cpol=*/0);
#else
  *(uint4*)l = *(const uint4*)g;
#endif
}

template <int N>
__device__ __forceinline__ void wait_async() {
#if HAS_ASYNC_LDS
#if __has_builtin(__builtin_amdgcn_s_wait_asynccnt)
  __builtin_amdgcn_s_wait_asynccnt((short)N);
#else
  asm volatile("s_wait_asynccnt %0" ::"n"(N));
#endif
#endif
}

// Issue one K-stage of copies: A tile 128x32 f32 (4 b128/thread),
// B tile 128x32 int8 (1 b128/thread). 5 async ops per wave per stage.
__device__ __forceinline__ void stage_copy(const float* __restrict__ x,
                                           const signed char* __restrict__ w,
                                           int m0, int n0, int kb,
                                           float* As, unsigned char* Bs, int tid) {
#pragma unroll
  for (int t = 0; t < 4; ++t) {
    int i = tid + t * 256;
    int row = i >> 3;        // 8 chunks of 4 f32 per row
    int ch  = i & 7;
    cp_b128(x + (size_t)(m0 + row) * GEMM_K + kb + ch * 4,
            As + row * A_STRIDE + ch * 4);
  }
  {
    int row = tid >> 1;      // 2 chunks of 16 B per row
    int ch  = tid & 1;
    cp_b128(w + (size_t)(n0 + row) * GEMM_K + kb + ch * 16,
            Bs + row * B_STRIDE + ch * 16);
  }
}

// A fragment (16x32 f16): lanes 0-15 hold K 0..15 of row M=lane,
// lanes 16-31 hold K 16..31 (matches B's contiguous order). f32->f16 via v_cvt_pk_f16_f32.
__device__ __forceinline__ v16h load_a_frag(const float* As, int row, int khalf) {
  const float* p = As + row * A_STRIDE + khalf * 16;
  float4 f0 = *(const float4*)(p + 0);
  float4 f1 = *(const float4*)(p + 4);
  float4 f2 = *(const float4*)(p + 8);
  float4 f3 = *(const float4*)(p + 12);
  v16h a;
  a[0]  = (_Float16)f0.x; a[1]  = (_Float16)f0.y; a[2]  = (_Float16)f0.z; a[3]  = (_Float16)f0.w;
  a[4]  = (_Float16)f1.x; a[5]  = (_Float16)f1.y; a[6]  = (_Float16)f1.z; a[7]  = (_Float16)f1.w;
  a[8]  = (_Float16)f2.x; a[9]  = (_Float16)f2.y; a[10] = (_Float16)f2.z; a[11] = (_Float16)f2.w;
  a[12] = (_Float16)f3.x; a[13] = (_Float16)f3.y; a[14] = (_Float16)f3.z; a[15] = (_Float16)f3.w;
  return a;
}

// B fragment (32x16 f16): lane holds 16 consecutive K of column N=lane%16
// (lanes 0-15: K 0..15, lanes 16-31: K 16..31) == 16 contiguous int8 of one
// W row. One ds_load_b128, then exact int8->f16 via the mantissa trick:
//   t = b ^ 0x80 (=b+128); f16 bits 0x6400|t == 1024+t == 1152+b; subtract 1152.
// 4 xor + 8 v_perm_b32 + 8 v_pk_add_f16 per fragment.
__device__ __forceinline__ v16h load_b_frag(const unsigned char* Bs, int row, int khalf) {
#if __has_builtin(__builtin_amdgcn_perm)
  union { uint32_t u[4]; int4 v; } raw;
  raw.v = *(const int4*)(Bs + row * B_STRIDE + khalf * 16);
  union { uint32_t u[8]; v16h h; } bh;
#pragma unroll
  for (int d = 0; d < 4; ++d) {
    uint32_t t = raw.u[d] ^ 0x80808080u;
    // dest bytes (LSB..MSB): [t.b0, 0x64, t.b1, 0x64] and [t.b2, 0x64, t.b3, 0x64]
    bh.u[2 * d + 0] = __builtin_amdgcn_perm(0x64646464u, t, 0x05010400u);
    bh.u[2 * d + 1] = __builtin_amdgcn_perm(0x64646464u, t, 0x05030402u);
  }
  v16h bias;
#pragma unroll
  for (int i = 0; i < 16; ++i) bias[i] = (_Float16)1152.0f;
  return bh.h - bias;
#else
  union { int4 v; signed char c[16]; } u;
  u.v = *(const int4*)(Bs + row * B_STRIDE + khalf * 16);
  v16h b;
#pragma unroll
  for (int i = 0; i < 16; ++i) b[i] = (_Float16)u.c[i];
  return b;
#endif
}

__global__ __launch_bounds__(256)
void WeightOnlyInt8Linear_wmma_kernel(const float* __restrict__ x,
                                      const signed char* __restrict__ w,
                                      const float* __restrict__ scales,
                                      float* __restrict__ out) {
  __shared__ float         As[2][BM * A_STRIDE];   // 36 KB
  __shared__ unsigned char Bs[2][BN * B_STRIDE];   // 12 KB

  const int tid   = threadIdx.x;
  const int lane  = tid & 31;
  const int wave  = tid >> 5;
  const int lrow  = lane & 15;
  const int khalf = lane >> 4;

  const int m0 = blockIdx.y * BM;
  const int n0 = blockIdx.x * BN;
  const int waveM = (wave >> 2) * 64;   // 2 wave-rows
  const int waveN = (wave & 3) * 32;    // 4 wave-cols

  v8f acc[4][2] = {};

  constexpr int STAGES = GEMM_K / KC;   // 128

  stage_copy(x, w, m0, n0, 0, As[0], Bs[0], tid);

  for (int s = 0; s < STAGES; ++s) {
    const int cur = s & 1;
    if (s + 1 < STAGES) {
      stage_copy(x, w, m0, n0, (s + 1) * KC, As[1 - cur], Bs[1 - cur], tid);
      wait_async<5>();   // stage s copies (5 oldest) retired; stage s+1 in flight
    } else {
      wait_async<0>();
    }
    __syncthreads();

    const float*         Ab = As[cur];
    const unsigned char* Bb = Bs[cur];

    v16h af[4];
#pragma unroll
    for (int mt = 0; mt < 4; ++mt)
      af[mt] = load_a_frag(Ab, waveM + mt * 16 + lrow, khalf);

    v16h bf[2];
#pragma unroll
    for (int nt = 0; nt < 2; ++nt)
      bf[nt] = load_b_frag(Bb, waveN + nt * 16 + lrow, khalf);

#pragma unroll
    for (int mt = 0; mt < 4; ++mt)
#pragma unroll
      for (int nt = 0; nt < 2; ++nt)
        acc[mt][nt] = __builtin_amdgcn_wmma_f32_16x16x32_f16(
            false, af[mt], false, bf[nt], (short)0, acc[mt][nt], false, false);

    __syncthreads();   // all waves done reading buf[cur] before it is refilled
  }

  // Epilogue: per-output-channel scale, C/D layout per ISA 7.12.2:
  // VGPR r: lanes 0-15 -> (M=r, N=lane); lanes 16-31 -> (M=r+8, N=lane-16)
#pragma unroll
  for (int nt = 0; nt < 2; ++nt) {
    const int col = n0 + waveN + nt * 16 + lrow;
    const float sc = scales[col];
#pragma unroll
    for (int mt = 0; mt < 4; ++mt) {
#pragma unroll
      for (int r = 0; r < 8; ++r) {
        const int rowg = m0 + waveM + mt * 16 + khalf * 8 + r;
        out[(size_t)rowg * GEMM_N + col] = acc[mt][nt][r] * sc;
      }
    }
  }
}

extern "C" void kernel_launch(void* const* d_in, const int* in_sizes, int n_in,
                              void* d_out, int out_size, void* d_ws, size_t ws_size,
                              hipStream_t stream) {
  const float*       x      = (const float*)d_in[0];
  const signed char* w      = (const signed char*)d_in[1];
  const float*       scales = (const float*)d_in[2];
  float*             out    = (float*)d_out;

  dim3 grid(GEMM_N / BN, GEMM_M / BM);   // 128 x 2
  dim3 block(256);
  hipLaunchKernelGGL(WeightOnlyInt8Linear_wmma_kernel, grid, block, 0, stream,
                     x, w, scales, out);
}
